// MultiHeadAttention_47364899340640
// MI455X (gfx1250) — matrix operand order repaired and checked by reading, not compiled, not run on hardware
//
#include <hip/hip_runtime.h>
#include <math.h>

// ---------------------------------------------------------------------------
// Problem constants (match reference)
// ---------------------------------------------------------------------------
static constexpr int Bb = 4;
static constexpr int Cc = 256;
static constexpr int Nn = 4096;
static constexpr int Mm = 4096;
static constexpr int Hh = 4;
static constexpr int Dd = 64;

// softmax(QK/sqrt(D)) computed in exp2 domain: fold 0.125*log2(e) into Q.
#define QSCALE 0.18033688011112042f   // 0.125 * log2(e)

typedef __bf16 bf16_t;
typedef __attribute__((ext_vector_type(16))) __bf16 v16bf;
typedef __attribute__((ext_vector_type(8)))  __bf16 v8bf;
typedef __attribute__((ext_vector_type(8)))  float  v8f;

union ABu { v16bf v; v8bf h[2]; };

// native v_exp_f32 (exp2)
__device__ __forceinline__ float exp2_fast(float x) {
  return __builtin_amdgcn_exp2f(x);
}

// ---------------------------------------------------------------------------
// WMMA fragment helpers (CDNA5 16x16x32 bf16, wave32)
//
// A-matrix (16 rows x 32 K), 16-bit layout (ISA 7.12.2):
//   lane L in [0,15]  : row L,  K = {0..7, 16..23}
//   lane L in [16,31] : row L-16, K = {8..15, 24..31}
// B-matrix (32 K x 16 cols):
//   lane L in [0,15]  : col L,  K = {0..7, 16..23}
//   lane L in [16,31] : col L-16, K = {8..15, 24..31}
// C/D (16x16 f32, 8 VGPRs): vgpr i, lane L -> row = i + 8*(L>=16), col = L%16
// ---------------------------------------------------------------------------
__device__ __forceinline__ v16bf load_frag_a(const bf16_t* base, int row_stride, int lane) {
  int r = lane & 15;
  int koff = (lane < 16) ? 0 : 8;
  const bf16_t* p = base + (size_t)r * row_stride + koff;
  ABu u;
  u.h[0] = *(const v8bf*)(p);        // K = koff .. koff+7
  u.h[1] = *(const v8bf*)(p + 16);   // K = koff+16 .. koff+23
  return u.v;
}

__device__ __forceinline__ v16bf load_frag_b(const bf16_t* base, int col_stride, int lane) {
  int c = lane & 15;
  int kbase = (lane < 16) ? 0 : 8;
  const bf16_t* p = base + (size_t)c * col_stride + kbase;
  ABu u;
  u.h[0] = *(const v8bf*)(p);        // K = kbase .. kbase+7
  u.h[1] = *(const v8bf*)(p + 16);   // K = kbase+16 .. kbase+23
  return u.v;
}

__device__ __forceinline__ v8f wmma_bf16(v16bf a, v16bf b, v8f c) {
  return __builtin_amdgcn_wmma_f32_16x16x32_bf16(
      /*neg_a=*/false, a, /*neg_b=*/false, b,
      /*c_mod=*/(short)0, c, /*reuse_a=*/false, /*reuse_b=*/false);
}

// Row-max reduction within a 16-lane half (rows replicated across the half)
__device__ __forceinline__ float half_reduce_max(float v) {
  v = fmaxf(v, __shfl_xor(v, 1, 32));
  v = fmaxf(v, __shfl_xor(v, 2, 32));
  v = fmaxf(v, __shfl_xor(v, 4, 32));
  v = fmaxf(v, __shfl_xor(v, 8, 32));
  return v;
}

// Online-softmax update for one 16x64 score tile (4 col-tiles in s[0..3]).
// Writes exp-weights P to LDS (row-major 16x64 at plrow), rescales acc[0..3].
__device__ __forceinline__ void softmax_tile(v8f* s, v8f* acc, float* mrow, float* corr,
                                             bf16_t* plrow, int half, int col) {
#pragma unroll
  for (int i = 0; i < 8; ++i) {
    float mx = fmaxf(fmaxf(s[0][i], s[1][i]), fmaxf(s[2][i], s[3][i]));
    mx = half_reduce_max(mx);
    float mnew = fmaxf(mrow[i], mx);
    corr[i] = exp2_fast(mrow[i] - mnew);
    mrow[i] = mnew;
    float e0 = exp2_fast(s[0][i] - mnew);
    float e1 = exp2_fast(s[1][i] - mnew);
    float e2 = exp2_fast(s[2][i] - mnew);
    float e3 = exp2_fast(s[3][i] - mnew);
    int r = i + half * 8;
    plrow[r * 64 + col]      = (bf16_t)e0;
    plrow[r * 64 + 16 + col] = (bf16_t)e1;
    plrow[r * 64 + 32 + col] = (bf16_t)e2;
    plrow[r * 64 + 48 + col] = (bf16_t)e3;
    acc[0][i] *= corr[i]; acc[1][i] *= corr[i];
    acc[2][i] *= corr[i]; acc[3][i] *= corr[i];
  }
}

// Re-read a 16x64 LDS P tile as two A fragments (m-chunks 0..31, 32..63)
__device__ __forceinline__ void load_pa(const bf16_t* plrow, int lane, v16bf& pa0, v16bf& pa1) {
  int r = lane & 15;
  int koff = (lane < 16) ? 0 : 8;
  ABu u0, u1;
  u0.h[0] = *(const v8bf*)(&plrow[r * 64 + koff]);
  u0.h[1] = *(const v8bf*)(&plrow[r * 64 + koff + 16]);
  u1.h[0] = *(const v8bf*)(&plrow[r * 64 + 32 + koff]);
  u1.h[1] = *(const v8bf*)(&plrow[r * 64 + 32 + koff + 16]);
  pa0 = u0.v;
  pa1 = u1.v;
}

// ---------------------------------------------------------------------------
// Kernel 0: convert the four weight matrices (C x C f32, row-major) to bf16.
// Wq additionally folds the attention scale + exp2 conversion factor.
// ---------------------------------------------------------------------------
__global__ void cvt_weights(const float* __restrict__ Wq, const float* __restrict__ Wk,
                            const float* __restrict__ Wv, const float* __restrict__ Wm,
                            bf16_t* __restrict__ out) {
  int i = blockIdx.x * blockDim.x + threadIdx.x;
  const int WN = Cc * Cc;
  if (i >= 4 * WN) return;
  int w = i / WN, j = i - w * WN;
  const float* src = (w == 0) ? Wq : (w == 1) ? Wk : (w == 2) ? Wv : Wm;
  float v = src[j];
  if (w == 0) v *= QSCALE;
  out[i] = (bf16_t)v;
}

// ---------------------------------------------------------------------------
// Kernel 1: transpose + convert  (B, C, L) f32  ->  (B, L, C) bf16
// block (32,8), grid (L/32, C/32, B)
// ---------------------------------------------------------------------------
__global__ void transpose_cvt(const float* __restrict__ x, bf16_t* __restrict__ xt, int L) {
  __shared__ float tile[32][33];
  int b  = blockIdx.z;
  int c0 = blockIdx.y * 32;
  int l0 = blockIdx.x * 32;
  int tx = threadIdx.x, ty = threadIdx.y;
#pragma unroll
  for (int i = ty; i < 32; i += 8)
    tile[i][tx] = x[(size_t)b * Cc * L + (size_t)(c0 + i) * L + (l0 + tx)];
  __syncthreads();
#pragma unroll
  for (int i = ty; i < 32; i += 8)
    xt[(size_t)b * L * Cc + (size_t)(l0 + i) * Cc + (c0 + tx)] = (bf16_t)tile[tx][i];
}

// ---------------------------------------------------------------------------
// Kernel 2: WMMA projection.  out[o,n] = sum_c W[o,c] * Xt[n,c] + bias[o]*bs
//   MODE 0: Q  -> bf16 (B,H,N,D)
//   MODE 1: K  -> bf16 (B,H,M,D)
//   MODE 2: V  -> bf16 (B,H,D,M)
//   MODE 3: Y  -> f32  (B,C,N)   (final output)
// block = 128 (4 waves, each a 16-row n-tile); grid = B*(C/16)*(L/64)
// ---------------------------------------------------------------------------
template <int MODE>
__global__ void proj_wmma(const bf16_t* __restrict__ Wb,
                          const float* __restrict__ bias,
                          float bias_scale,
                          const bf16_t* __restrict__ Xt,  // (B, L, C) bf16
                          bf16_t* __restrict__ out_bf,
                          float* __restrict__ out_f,
                          int L) {
  const int lane = threadIdx.x & 31;
  const int wave = threadIdx.x >> 5;
  int tile = blockIdx.x;
  const int OT = Cc / 16;     // 16 o-tiles
  const int NB = L / 64;      // n super-blocks (64 rows per block)
  int ot = tile % OT; tile /= OT;
  int nb = tile % NB; int b = tile / NB;
  int o0 = ot * 16;
  int n0 = nb * 64 + wave * 16;

  const bf16_t* Wrow = Wb + (size_t)o0 * Cc;
  const bf16_t* Xrow = Xt + ((size_t)b * L + n0) * Cc;

  v8f acc = {};
#pragma unroll
  for (int c0 = 0; c0 < Cc; c0 += 32) {
    v16bf af = load_frag_a(Wrow + c0, Cc, lane);
    v16bf bfg = load_frag_b(Xrow + c0, Cc, lane);
    acc = wmma_bf16(af, bfg, acc);
  }

  const int half = lane >> 4;
  const int col  = lane & 15;
#pragma unroll
  for (int i = 0; i < 8; ++i) {
    int o = o0 + i + half * 8;
    int n = n0 + col;
    float v = acc[i] + bias[o] * bias_scale;
    if (MODE == 0 || MODE == 1) {
      int h = o >> 6, d = o & 63;
      out_bf[(((size_t)b * Hh + h) * (size_t)L + n) * Dd + d] = (bf16_t)v;
    } else if (MODE == 2) {
      int h = o >> 6, d = o & 63;
      out_bf[(((size_t)b * Hh + h) * (size_t)Dd + d) * (size_t)L + n] = (bf16_t)v;
    } else {
      out_f[((size_t)b * Cc + o) * (size_t)L + n] = v;
    }
  }
}

// ---------------------------------------------------------------------------
// Kernel 3: fused flash attention, 32 query rows per wave (two 16-row tiles
// A and B) so every K/V B-fragment load feeds two WMMAs.
// __launch_bounds__(128, 1): allow full VGPR budget (no spills) — each wave
// carries 8+ independent WMMA chains, so 1-2 waves/SIMD suffices for latency.
//   Qb: (B,H,N,D) bf16 (pre-scaled by 0.125*log2e)
//   Kb: (B,H,M,D) bf16   Vb: (B,H,D,M) bf16
//   attT: (B,N,C) bf16   (C index = h*64 + d, ready for final projection)
// block = 128 (4 waves -> 128 query rows); grid = B*H*(N/128)
// ---------------------------------------------------------------------------
__global__ void __launch_bounds__(128, 1)
attn_wmma(const bf16_t* __restrict__ Qb,
          const bf16_t* __restrict__ Kb,
          const bf16_t* __restrict__ Vb,
          bf16_t* __restrict__ attT) {
  __shared__ bf16_t plds[4 * 32 * 64];  // per-wave 32x64 P staging (16 KB)
  const int lane = threadIdx.x & 31;
  const int wave = threadIdx.x >> 5;
  int blk = blockIdx.x;
  const int NB = Nn / 128;
  int nb = blk % NB; blk /= NB;
  int h = blk % Hh; int b = blk / Hh;
  int n0 = nb * 128 + wave * 32;

  const bf16_t* Qp    = Qb + (((size_t)b * Hh + h) * Nn + n0) * Dd;
  const bf16_t* Kbase = Kb + ((size_t)b * Hh + h) * (size_t)Mm * Dd;
  const bf16_t* Vbase = Vb + ((size_t)b * Hh + h) * (size_t)Dd * Mm;
  bf16_t* plA = plds + wave * (32 * 64);
  bf16_t* plB = plA + 16 * 64;

  // Q tiles resident in registers: rows n0..n0+15 (A), n0+16..n0+31 (B)
  v16bf qaA0 = load_frag_a(Qp, Dd, lane);
  v16bf qaA1 = load_frag_a(Qp + 32, Dd, lane);
  v16bf qaB0 = load_frag_a(Qp + 16 * Dd, Dd, lane);
  v16bf qaB1 = load_frag_a(Qp + 16 * Dd + 32, Dd, lane);

  // constant all-ones B fragment for row-sum WMMA
  v16bf onesb;
#pragma unroll
  for (int i = 0; i < 16; ++i) onesb[i] = (bf16_t)1.0f;

  v8f accA[4] = {{}, {}, {}, {}};
  v8f accB[4] = {{}, {}, {}, {}};
  float mrowA[8], lrowA[8], mrowB[8], lrowB[8];
#pragma unroll
  for (int i = 0; i < 8; ++i) {
    mrowA[i] = -1e30f; lrowA[i] = 0.f;
    mrowB[i] = -1e30f; lrowB[i] = 0.f;
  }

  const int half = lane >> 4;
  const int col  = lane & 15;

  for (int m0 = 0; m0 < Mm; m0 += 64) {
    const bf16_t* Kp = Kbase + (size_t)m0 * Dd;
    if (m0 + 64 < Mm) {
      __builtin_prefetch(Kbase + (size_t)(m0 + 64) * Dd, 0, 1);   // K ahead
      __builtin_prefetch(Vbase + (size_t)(m0 + 64), 0, 1);        // V ahead
    }

    // S = Q * K^T : four 16x16 col-tiles per row-tile; each K fragment
    // feeds both row-tiles.
    v8f sA[4], sB[4];
#pragma unroll
    for (int t = 0; t < 4; ++t) {
      v16bf kb0 = load_frag_b(Kp + (size_t)t * 16 * Dd, Dd, lane);
      v16bf kb1 = load_frag_b(Kp + (size_t)t * 16 * Dd + 32, Dd, lane);
      v8f z0 = {}, z1 = {};
      sA[t] = wmma_bf16(qaA1, kb1, wmma_bf16(qaA0, kb0, z0));
      sB[t] = wmma_bf16(qaB1, kb1, wmma_bf16(qaB0, kb0, z1));
    }

    // online softmax (exp2 domain, scale pre-folded into Q)
    float corrA[8], corrB[8];
    softmax_tile(sA, accA, mrowA, corrA, plA, half, col);
    softmax_tile(sB, accB, mrowB, corrB, plB, half, col);

    // re-read P as A fragments
    v16bf paA0, paA1, paB0, paB1;
    load_pa(plA, lane, paA0, paA1);
    load_pa(plB, lane, paB0, paB1);

    // row sums of P via WMMA against ones (columns replicated => per-row sum)
    {
      v8f z0 = {}, z1 = {};
      v8f rsA = wmma_bf16(paA1, onesb, wmma_bf16(paA0, onesb, z0));
      v8f rsB = wmma_bf16(paB1, onesb, wmma_bf16(paB0, onesb, z1));
#pragma unroll
      for (int i = 0; i < 8; ++i) {
        lrowA[i] = lrowA[i] * corrA[i] + rsA[i];
        lrowB[i] = lrowB[i] * corrB[i] + rsB[i];
      }
    }

    // O += P * V^T : four d-tiles, two m-chunks; each V fragment feeds both
    // row-tiles.
#pragma unroll
    for (int j = 0; j < 4; ++j) {
      v16bf vb0 = load_frag_b(Vbase + (size_t)j * 16 * Mm + m0, Mm, lane);
      v16bf vb1 = load_frag_b(Vbase + (size_t)j * 16 * Mm + m0 + 32, Mm, lane);
      accA[j] = wmma_bf16(paA1, vb1, wmma_bf16(paA0, vb0, accA[j]));
      accB[j] = wmma_bf16(paB1, vb1, wmma_bf16(paB0, vb0, accB[j]));
    }
  }

  // normalize and store to attT (B, N, C) with C index = h*64 + d
#pragma unroll
  for (int i = 0; i < 8; ++i) {
    int rA = n0 + i + half * 8;
    int rB = rA + 16;
    float invA = 1.0f / lrowA[i];
    float invB = 1.0f / lrowB[i];
    bf16_t* dstA = attT + ((size_t)b * Nn + rA) * Cc + h * 64;
    bf16_t* dstB = attT + ((size_t)b * Nn + rB) * Cc + h * 64;
#pragma unroll
    for (int j = 0; j < 4; ++j) {
      dstA[j * 16 + col] = (bf16_t)(accA[j][i] * invA);
      dstB[j * 16 + col] = (bf16_t)(accB[j][i] * invB);
    }
  }
}

// ---------------------------------------------------------------------------
// Host-side launch
// ---------------------------------------------------------------------------
extern "C" void kernel_launch(void* const* d_in, const int* in_sizes, int n_in,
                              void* d_out, int out_size, void* d_ws, size_t ws_size,
                              hipStream_t stream) {
  const float* query  = (const float*)d_in[0];
  const float* source = (const float*)d_in[1];
  const float* Wq = (const float*)d_in[2];
  const float* bq = (const float*)d_in[3];
  const float* Wk = (const float*)d_in[4];
  const float* bk = (const float*)d_in[5];
  const float* Wv = (const float*)d_in[6];
  const float* bv = (const float*)d_in[7];
  const float* Wm = (const float*)d_in[8];
  const float* bm = (const float*)d_in[9];
  float* out = (float*)d_out;

  char* ws = (char*)d_ws;
  size_t off = 0;
  bf16_t* Wall = (bf16_t*)(ws + off); off += (size_t)4 * Cc * Cc * sizeof(bf16_t);
  bf16_t* Wq_b = Wall + 0 * Cc * Cc;
  bf16_t* Wk_b = Wall + 1 * Cc * Cc;
  bf16_t* Wv_b = Wall + 2 * Cc * Cc;
  bf16_t* Wm_b = Wall + 3 * Cc * Cc;
  bf16_t* qt   = (bf16_t*)(ws + off); off += (size_t)Bb * Nn * Cc * sizeof(bf16_t);
  bf16_t* st   = (bf16_t*)(ws + off); off += (size_t)Bb * Mm * Cc * sizeof(bf16_t);
  bf16_t* Qb   = (bf16_t*)(ws + off); off += (size_t)Bb * Hh * Nn * Dd * sizeof(bf16_t);
  bf16_t* Kb   = (bf16_t*)(ws + off); off += (size_t)Bb * Hh * Mm * Dd * sizeof(bf16_t);
  bf16_t* Vb   = (bf16_t*)(ws + off); off += (size_t)Bb * Hh * Dd * Mm * sizeof(bf16_t);
  bf16_t* attT = (bf16_t*)(ws + off); off += (size_t)Bb * Nn * Cc * sizeof(bf16_t);
  (void)ws_size; (void)in_sizes; (void)n_in; (void)out_size;

  // 0) weights -> bf16 (Wq pre-scaled by 0.125*log2e)
  {
    int total = 4 * Cc * Cc;
    cvt_weights<<<(total + 255) / 256, 256, 0, stream>>>(Wq, Wk, Wv, Wm, Wall);
  }
  // 1) transpose+convert activations
  {
    dim3 blk(32, 8);
    transpose_cvt<<<dim3(Nn / 32, Cc / 32, Bb), blk, 0, stream>>>(query,  qt, Nn);
    transpose_cvt<<<dim3(Mm / 32, Cc / 32, Bb), blk, 0, stream>>>(source, st, Mm);
  }
  // 2) Q / K / V projections (WMMA)
  {
    int grid = Bb * (Cc / 16) * (Nn / 64);
    proj_wmma<0><<<grid, 128, 0, stream>>>(Wq_b, bq, QSCALE, qt, Qb, nullptr, Nn);
    grid = Bb * (Cc / 16) * (Mm / 64);
    proj_wmma<1><<<grid, 128, 0, stream>>>(Wk_b, bk, 1.0f, st, Kb, nullptr, Mm);
    proj_wmma<2><<<grid, 128, 0, stream>>>(Wv_b, bv, 1.0f, st, Vb, nullptr, Mm);
  }
  // 3) fused flash attention (WMMA, 32 rows/wave)
  {
    int grid = Bb * Hh * (Nn / 128);
    attn_wmma<<<grid, 128, 0, stream>>>(Qb, Kb, Vb, attT);
  }
  // 4) output projection (WMMA, f32 out)
  {
    int grid = Bb * (Cc / 16) * (Nn / 64);
    proj_wmma<3><<<grid, 128, 0, stream>>>(Wm_b, bm, 1.0f, attT, nullptr, out, Nn);
  }
}
